// Actor_5188320494285
// MI455X (gfx1250) — matrix operand order repaired and checked by reading, not compiled, not run on hardware
//
#include <hip/hip_runtime.h>
#include <hip/hip_bf16.h>
#include <math.h>

// Problem constants
#define BATCH   65536
#define DIN     376
#define DPAD    384
#define HID     256
#define NACT    17
#define NEXP    8
#define TILE_M  32
#define CAP     (BATCH + NEXP * TILE_M)       // padded index buffer capacity
#define NTILES  (BATCH / TILE_M + NEXP)       // 2056 fixed grid
#define LDH     264                            // LDS row stride in halves

#define LOG_STD_MAX 2.0f
#define LOG_STD_MIN (-5.0f)

typedef __attribute__((ext_vector_type(16))) _Float16 v16h;
typedef __attribute__((ext_vector_type(8)))  _Float16 v8h;
typedef __attribute__((ext_vector_type(8)))  float    v8f;

// ---------------------------------------------------------------------------
// Prep kernels: expert bucketing + weight repack to f16 B-fragment layout
// ---------------------------------------------------------------------------
__global__ void init_kernel(int* __restrict__ idxbuf, int* __restrict__ counts) {
    int i = blockIdx.x * blockDim.x + threadIdx.x;
    if (i < CAP)  idxbuf[i] = -1;      // sentinel row
    if (i < NEXP) counts[i] = 0;
}

__global__ void count_kernel(const long long* __restrict__ o, int* __restrict__ counts) {
    int i = blockIdx.x * blockDim.x + threadIdx.x;
    if (i < BATCH) atomicAdd(&counts[(int)o[i]], 1);
}

__global__ void plan_kernel(const int* __restrict__ counts, int* __restrict__ cursors,
                            int* __restrict__ tile_expert) {
    if (blockIdx.x == 0 && threadIdx.x == 0) {
        int off = 0, tile = 0;
        for (int e = 0; e < NEXP; ++e) {
            cursors[e] = off;
            int tiles = (counts[e] + TILE_M - 1) / TILE_M;
            for (int i = 0; i < tiles; ++i) tile_expert[tile++] = e;
            off += tiles * TILE_M;
        }
        for (; tile < NTILES; ++tile) tile_expert[tile] = -1;
    }
}

__global__ void scatter_kernel(const long long* __restrict__ o, int* __restrict__ cursors,
                               int* __restrict__ idxbuf) {
    int i = blockIdx.x * blockDim.x + threadIdx.x;
    if (i < BATCH) {
        int e = (int)o[i];
        int pos = atomicAdd(&cursors[e], 1);
        idxbuf[pos] = i;
    }
}

// Wt[n][k] layouts (f16, K contiguous) so a B fragment is two 16B loads per lane.
__global__ void wconv_kernel(const float* __restrict__ W1, const float* __restrict__ W2,
                             const float* __restrict__ Wm, const float* __restrict__ Ws,
                             _Float16* __restrict__ W1t, _Float16* __restrict__ W2t,
                             _Float16* __restrict__ Wht) {
    const int n1 = NEXP * HID * DPAD;   // 786432
    const int n2 = NEXP * HID * HID;    // 524288
    const int nh = NEXP * 64 * HID;     // 131072
    int i = blockIdx.x * blockDim.x + threadIdx.x;
    if (i < n1) {
        int k = i % DPAD; int h = (i / DPAD) % HID; int e = i / (DPAD * HID);
        float v = (k < DIN) ? W1[((size_t)e * DIN + k) * HID + h] : 0.f;
        W1t[i] = (_Float16)v;
    } else if (i < n1 + n2) {
        int j = i - n1;
        int k = j % HID; int n = (j / HID) % HID; int e = j / (HID * HID);
        W2t[j] = (_Float16)W2[((size_t)e * HID + k) * HID + n];
    } else if (i < n1 + n2 + nh) {
        int j = i - n1 - n2;
        int k = j % HID; int n = (j / HID) % 64; int e = j / (64 * HID);
        float v = 0.f;
        if (n < 32) { if (n < NACT) v = Wm[((size_t)e * HID + k) * NACT + n]; }
        else        { int a = n - 32; if (a < NACT) v = Ws[((size_t)e * HID + k) * NACT + a]; }
        Wht[j] = (_Float16)v;
    }
}

// ---------------------------------------------------------------------------
// Fragment loaders (CDNA5 16-bit WMMA layouts, §7.12.2)
// ---------------------------------------------------------------------------
// A 16x32: lane holds row M = lane&15; halves [kb+kg*8 .. +8) -> a[0..7],
//          halves [kb+16+kg*8 .. +8) -> a[8..15], kg = lane>>4.
__device__ __forceinline__ v16h load_a_f32(const float* __restrict__ rowp, int kb, int kg) {
    v16h a;
    int c0 = kb + kg * 8, c1 = kb + 16 + kg * 8;
#pragma unroll
    for (int i = 0; i < 8; ++i) {
        a[i]     = (_Float16)rowp[c0 + i];
        a[i + 8] = (_Float16)rowp[c1 + i];
    }
    return a;
}
__device__ __forceinline__ v16h load_a_f32_tail(const float* __restrict__ rowp, int kb, int kg) {
    v16h a;
    int c0 = kb + kg * 8, c1 = kb + 16 + kg * 8;
#pragma unroll
    for (int i = 0; i < 8; ++i) {
        a[i]     = (_Float16)((c0 + i < DIN) ? rowp[c0 + i] : 0.f);
        a[i + 8] = (_Float16)((c1 + i < DIN) ? rowp[c1 + i] : 0.f);
    }
    return a;
}
__device__ __forceinline__ v16h load_a_lds(const _Float16* hrow, int kb, int kg) {
    v16h a;
    v8h c0 = *(const v8h*)(hrow + kb + kg * 8);
    v8h c1 = *(const v8h*)(hrow + kb + 16 + kg * 8);
#pragma unroll
    for (int i = 0; i < 8; ++i) { a[i] = c0[i]; a[i + 8] = c1[i]; }
    return a;
}
// B 32x16: lane holds col N = lane&15, halves Wt[n][kb + kg*8 .. +8] -> b[0..7],
//          Wt[n][kb + 16 + kg*8 .. +8] -> b[8..15], kg = lane>>4.
__device__ __forceinline__ v16h load_b(const _Float16* __restrict__ wrow, int kb, int kg) {
    v16h b;
    v8h c0 = *(const v8h*)(wrow + kb + kg * 8);
    v8h c1 = *(const v8h*)(wrow + kb + 16 + kg * 8);
#pragma unroll
    for (int i = 0; i < 8; ++i) { b[i] = c0[i]; b[i + 8] = c1[i]; }
    return b;
}

// ---------------------------------------------------------------------------
// Fused per-tile MoE actor MLP (32 rows, one expert per tile)
// ---------------------------------------------------------------------------
__global__ __launch_bounds__(256) void moe_actor_kernel(
    const float* __restrict__ x,
    const float* __restrict__ b1, const float* __restrict__ b2,
    const float* __restrict__ bm, const float* __restrict__ bs,
    const _Float16* __restrict__ W1t, const _Float16* __restrict__ W2t,
    const _Float16* __restrict__ Wht,
    const int* __restrict__ idxbuf, const int* __restrict__ tile_expert,
    float* __restrict__ out)
{
    __shared__ int       ridx[TILE_M];
    __shared__ _Float16  h1[TILE_M * LDH];
    __shared__ _Float16  h2[TILE_M * LDH];

    const int t = blockIdx.x;
    const int e = tile_expert[t];
    if (e < 0) return;

    const int tid = threadIdx.x;
    if (tid < TILE_M) ridx[tid] = idxbuf[t * TILE_M + tid];
    __syncthreads();

    const int lane = tid & 31;
    const int wave = tid >> 5;
    const int wm   = wave >> 2;      // 0..1 : 16-row band
    const int wn   = wave & 3;       // 0..3 : 16-col band
    const int lrow = lane & 15;
    const int kg   = lane >> 4;

    // ---------------- Layer 1: h1 = relu(x @ W1[e] + b1[e]) ----------------
    const int  r0 = ridx[wm * 16 + lrow];
    const float* xr = x + (size_t)(r0 < 0 ? 0 : r0) * DIN;
    const _Float16* w1e = W1t + (size_t)e * HID * DPAD;

    v8f acc[4];
#pragma unroll
    for (int p = 0; p < 4; ++p) acc[p] = (v8f){0.f,0.f,0.f,0.f,0.f,0.f,0.f,0.f};

#pragma unroll
    for (int kb = 0; kb < 352; kb += 32) {
        v16h a = load_a_f32(xr, kb, kg);
#pragma unroll
        for (int p = 0; p < 4; ++p) {
            int ncol = wn * 16 + p * 64 + lrow;
            v16h b = load_b(w1e + (size_t)ncol * DPAD, kb, kg);
            acc[p] = __builtin_amdgcn_wmma_f32_16x16x32_f16(false, a, false, b,
                                                            (short)0, acc[p], false, false);
        }
    }
    {   // K tail 352..383 (valid to 375; W1t zero-padded)
        v16h a = load_a_f32_tail(xr, 352, kg);
#pragma unroll
        for (int p = 0; p < 4; ++p) {
            int ncol = wn * 16 + p * 64 + lrow;
            v16h b = load_b(w1e + (size_t)ncol * DPAD, 352, kg);
            acc[p] = __builtin_amdgcn_wmma_f32_16x16x32_f16(false, a, false, b,
                                                            (short)0, acc[p], false, false);
        }
    }
#pragma unroll
    for (int p = 0; p < 4; ++p) {
        int ncol = wn * 16 + p * 64 + lrow;
        float bias = b1[e * HID + ncol];
#pragma unroll
        for (int r = 0; r < 8; ++r) {
            float v = acc[p][r] + bias;
            v = v > 0.f ? v : 0.f;
            int m = wm * 16 + kg * 8 + r;
            h1[m * LDH + ncol] = (_Float16)v;
        }
    }
    __syncthreads();

    // ---------------- Layer 2: h2 = relu(h1 @ W2[e] + b2[e]) ----------------
    const _Float16* w2e = W2t + (size_t)e * HID * HID;
    const _Float16* hA1 = &h1[(wm * 16 + lrow) * LDH];
#pragma unroll
    for (int p = 0; p < 4; ++p) acc[p] = (v8f){0.f,0.f,0.f,0.f,0.f,0.f,0.f,0.f};
#pragma unroll
    for (int kb = 0; kb < HID; kb += 32) {
        v16h a = load_a_lds(hA1, kb, kg);
#pragma unroll
        for (int p = 0; p < 4; ++p) {
            int ncol = wn * 16 + p * 64 + lrow;
            v16h b = load_b(w2e + (size_t)ncol * HID, kb, kg);
            acc[p] = __builtin_amdgcn_wmma_f32_16x16x32_f16(false, a, false, b,
                                                            (short)0, acc[p], false, false);
        }
    }
#pragma unroll
    for (int p = 0; p < 4; ++p) {
        int ncol = wn * 16 + p * 64 + lrow;
        float bias = b2[e * HID + ncol];
#pragma unroll
        for (int r = 0; r < 8; ++r) {
            float v = acc[p][r] + bias;
            v = v > 0.f ? v : 0.f;
            int m = wm * 16 + kg * 8 + r;
            h2[m * LDH + ncol] = (_Float16)v;
        }
    }
    __syncthreads();

    // ------- Heads: [mean | log_std] = h2 @ Wht[e] (64 combined cols) -------
    const _Float16* whe = Wht + (size_t)e * 64 * HID;
    const _Float16* hA2 = &h2[(wm * 16 + lrow) * LDH];
    const int ncol = wn * 16 + lrow;            // 0..63 combined head column
    v8f ah = (v8f){0.f,0.f,0.f,0.f,0.f,0.f,0.f,0.f};
#pragma unroll
    for (int kb = 0; kb < HID; kb += 32) {
        v16h a = load_a_lds(hA2, kb, kg);
        v16h b = load_b(whe + (size_t)ncol * HID, kb, kg);
        ah = __builtin_amdgcn_wmma_f32_16x16x32_f16(false, a, false, b,
                                                    (short)0, ah, false, false);
    }

    const bool is_std = (wn >= 2);
    const int  a_idx  = (wn & 1) * 16 + lrow;   // wn 0/2 -> 0..15, wn 1/3 -> 16..31
    if (a_idx < NACT) {
        float bias = is_std ? bs[e * NACT + a_idx] : bm[e * NACT + a_idx];
#pragma unroll
        for (int r = 0; r < 8; ++r) {
            int m   = wm * 16 + kg * 8 + r;
            int row = ridx[m];
            if (row >= 0) {
                float v = ah[r] + bias;
                if (is_std) {
                    float tt = tanhf(v);
                    v = LOG_STD_MIN + 0.5f * (LOG_STD_MAX - LOG_STD_MIN) * (tt + 1.f);
                    out[(size_t)BATCH * NACT + (size_t)row * NACT + a_idx] = v;
                } else {
                    out[(size_t)row * NACT + a_idx] = v;
                }
            }
        }
    }
}

// ---------------------------------------------------------------------------
extern "C" void kernel_launch(void* const* d_in, const int* in_sizes, int n_in,
                              void* d_out, int out_size, void* d_ws, size_t ws_size,
                              hipStream_t stream) {
    const float*     x  = (const float*)d_in[0];
    const long long* o  = (const long long*)d_in[1];
    const float*     W1 = (const float*)d_in[2];
    const float*     b1 = (const float*)d_in[3];
    const float*     W2 = (const float*)d_in[4];
    const float*     b2 = (const float*)d_in[5];
    const float*     Wm = (const float*)d_in[6];
    const float*     bm = (const float*)d_in[7];
    const float*     Ws = (const float*)d_in[8];
    const float*     bs = (const float*)d_in[9];
    float* out = (float*)d_out;

    // Workspace layout (all offsets 512B-aligned for 16B vector loads)
    char* ws = (char*)d_ws;
    size_t off = 0;
    int* counts      = (int*)(ws + off); off += 512;
    int* cursors     = (int*)(ws + off); off += 512;
    int* tile_expert = (int*)(ws + off); off += ((size_t)NTILES * 4 + 511) / 512 * 512;
    int* idxbuf      = (int*)(ws + off); off += ((size_t)CAP * 4 + 511) / 512 * 512;
    _Float16* W1t    = (_Float16*)(ws + off); off += (size_t)NEXP * HID * DPAD * 2;
    _Float16* W2t    = (_Float16*)(ws + off); off += (size_t)NEXP * HID * HID * 2;
    _Float16* Wht    = (_Float16*)(ws + off); off += (size_t)NEXP * 64 * HID * 2;

    init_kernel   <<<(CAP + 255) / 256, 256, 0, stream>>>(idxbuf, counts);
    count_kernel  <<<(BATCH + 255) / 256, 256, 0, stream>>>(o, counts);
    plan_kernel   <<<1, 32, 0, stream>>>(counts, cursors, tile_expert);
    scatter_kernel<<<(BATCH + 255) / 256, 256, 0, stream>>>(o, cursors, idxbuf);
    {
        const int nconv = NEXP * HID * DPAD + NEXP * HID * HID + NEXP * 64 * HID;
        wconv_kernel<<<(nconv + 255) / 256, 256, 0, stream>>>(W1, W2, Wm, Ws, W1t, W2t, Wht);
    }
    moe_actor_kernel<<<NTILES, 256, 0, stream>>>(x, b1, b2, bm, bs,
                                                 W1t, W2t, Wht, idxbuf, tile_expert, out);
}
